// Weighted_Gaussian_74294344286374
// MI455X (gfx1250) — matrix operand order repaired and checked by reading, not compiled, not run on hardware
//
#include <hip/hip_runtime.h>
#include <math.h>

typedef __attribute__((ext_vector_type(16))) _Float16 v16h;
typedef __attribute__((ext_vector_type(8)))  _Float16 v8h;
typedef __attribute__((ext_vector_type(8)))  float    v8f;
typedef __attribute__((ext_vector_type(4)))  unsigned int v4u;
typedef __attribute__((ext_vector_type(8)))  int      v8i;
typedef __attribute__((ext_vector_type(4)))  int      v4i;

__device__ __forceinline__ v8f wmma16x16x32(v16h a, v16h b, v8f c) {
  return __builtin_amdgcn_wmma_f32_16x16x32_f16(false, a, false, b, (short)0, c,
                                                false, false);
}

// B-matrix 32x16 k map (lane holds col N=lane%16; lanes<16 K0..7,16..23, else K8..15,24..31)
__device__ __forceinline__ int b_kmap(int lane, int j) {
  return ((lane & 16) ? 8 : 0) + j + ((j >= 8) ? 8 : 0);
}

// Conv3d A fragment from contiguous f16 LDS slab [9][11][26]; TB = 32*b block
// base, KB = lane-half base (template constants -> t, kh, kw fold to immediate
// ds_load_u16 offsets; non-linear t->(kh,kw) forces the per-half split).
template <int TB, int KB>
__device__ __forceinline__ v16h build_A_c3(const _Float16* __restrict__ sl) {
  v16h A;
#pragma unroll
  for (int j = 0; j < 16; ++j) {
    const int t = TB + KB + j + ((j >= 8) ? 8 : 0);
    if (t < 121) {
      const int kh = t / 11;
      const int kw = t - kh * 11;
      A[j] = sl[kh * 26 + kw];
    } else {
      A[j] = (_Float16)0.0f;
    }
  }
  return A;
}

// Conv2d A fragment: offset is LINEAR in k, so the lane-half base (8*54) is
// folded into `sl` by the caller -> one branch-free arm, compile-time offsets.
__device__ __forceinline__ v16h build_A_c2(const _Float16* __restrict__ sl) {
  v16h A;
#pragma unroll
  for (int j = 0; j < 16; ++j) {
    const int jm = j + ((j >= 8) ? 8 : 0);
    A[j] = sl[jm * 54];
  }
  return A;
}

// MLP A fragment: row-major by k -> two contiguous aligned 8-half chunks.
template <int K0>
__device__ __forceinline__ v16h build_A_act(const _Float16* __restrict__ row,
                                            int kb) {
  const v8h lo = *(const v8h*)(row + K0 + kb);
  const v8h hi = *(const v8h*)(row + K0 + kb + 16);
  v16h A;
#pragma unroll
  for (int j = 0; j < 8; ++j) { A[j] = lo[j]; A[j + 8] = hi[j]; }
  return A;
}

// ---------------------------------------------------------------------------
// Setup kernels
// ---------------------------------------------------------------------------
__global__ __launch_bounds__(256) void setup_Bc2(const float* __restrict__ W3,
                                                 _Float16* __restrict__ dst) {
  const int i = blockIdx.x * 256 + threadIdx.x;           // 1296*512
  if (i >= 1296 * 512) return;
  const int f = i >> 9, r = i & 511, lane = r >> 4, j = r & 15;
  const int nt = f % 12, rem = f / 12, icb = rem % 12, tap = rem / 12;
  const int ic = icb * 32 + b_kmap(lane, j);
  const int c = nt * 16 + (lane & 15);
  dst[i] = (_Float16)W3[c * 3456 + ic * 9 + tap];
}

__global__ __launch_bounds__(256) void setup_Bc3(const float* __restrict__ W1,
                                                 _Float16* __restrict__ dst) {
  const int i = blockIdx.x * 256 + threadIdx.x;           // 44*512
  if (i >= 44 * 512) return;
  const int f = i >> 9, r = i & 511, lane = r >> 4, j = r & 15;
  const int nt = f & 1, step = f >> 1;                    // 22 k-steps
  const int kd = (step < 20) ? (step >> 2) : 5;
  const int b  = (step < 20) ? (step & 3) : (step - 20);
  const int t = b * 32 + b_kmap(lane, j);
  const int c = nt * 16 + (lane & 15);
  const bool keep = (t < 121) && ((kd < 5) || (t < 60)) && (c < 24);
  dst[i] = (_Float16)(keep ? W1[c * 1331 + kd * 121 + t] : 0.0f);
}

__global__ __launch_bounds__(256) void setup_Bmlp(const float* __restrict__ Wa,
                                                  const float* __restrict__ Wb,
                                                  const float* __restrict__ Wc,
                                                  _Float16* __restrict__ dst) {
  const int i = blockIdx.x * 256 + threadIdx.x;           // 18*512
  if (i >= 18 * 512) return;
  const int f = i >> 9, r = i & 511, lane = r >> 4, j = r & 15;
  const int n = lane & 15;
  const int kmap = b_kmap(lane, j);
  float v = 0.0f;
  if (f < 3) {
    const int k = kmap;
    if (k < 25) v = Wa[(f * 16 + n) * 25 + k];
  } else if (f < 15) {
    const int g = f - 3, ks = g / 6, ntl = g % 6;
    const int k = ks * 32 + kmap;
    if (k < 48) v = Wb[(ntl * 16 + n) * 48 + k];
  } else {
    const int k = (f - 15) * 32 + kmap;
    if (n < 9) v = Wc[n * 96 + k];
  }
  dst[i] = (_Float16)v;
}

__global__ __launch_bounds__(256) void setup_x16(const float* __restrict__ x,
                                                 _Float16* __restrict__ x16) {
  const int i = blockIdx.x * 256 + threadIdx.x;           // 442368
  if (i < 442368) x16[i] = (_Float16)x[i];
}

// ---------------------------------------------------------------------------
// Kernel 1: conv2d via implicit GEMM. Block stages its hyper patch in LDS as
// f16 once (zero-padded); A fragments are branch-free ds_load_u16 with
// immediate offsets; wave w owns N-tiles 3w..3w+2.
// ---------------------------------------------------------------------------
__global__ __launch_bounds__(128) void conv2d_hyper_wmma(
    const float* __restrict__ hyper, const _Float16* __restrict__ wsBc2,
    const float* __restrict__ b3, float* __restrict__ hout)
{
  __shared__ _Float16 sh2[384 * 3 * 18];               // 41.5 KB

  const int lane  = threadIdx.x & 31;
  const int wave  = threadIdx.x >> 5;
  const int sp0   = blockIdx.x * 16;                   // 144 Mtiles
  const int yy    = sp0 / 48;
  const int x0    = sp0 % 48;
  const int am    = lane & 15;
  const int ahalf = (lane >> 4) & 1;
  const int nt0   = wave * 3;

  for (int i = threadIdx.x; i < 384 * 3 * 18; i += 128) {
    const int ic = i / 54;
    const int rr = i - ic * 54;
    const int ky = rr / 18, col = rr - ky * 18;
    const int hy = yy - 1 + ky, wx = x0 - 1 + col;
    float v = 0.0f;
    if (hy >= 0 && hy < 48 && wx >= 0 && wx < 48)
      v = hyper[ic * 2304 + hy * 48 + wx];
    sh2[i] = (_Float16)v;
  }
  __syncthreads();

  v8f acc[3];
#pragma unroll
  for (int q = 0; q < 3; ++q) {
    const float bi = b3[(nt0 + q) * 16 + (lane & 15)];
#pragma unroll
    for (int r = 0; r < 8; ++r) acc[q][r] = bi;
  }

  for (int tap = 0; tap < 9; ++tap) {
    const int ky = tap / 3, kx = tap % 3;
    // lane-half k-base (8*54) folded into the base pointer: branch-free build
    const _Float16* stap = sh2 + ky * 18 + am + kx + ahalf * (8 * 54);
    for (int icb = 0; icb < 12; ++icb) {
      const _Float16* sl = stap + icb * (32 * 54);
      const v16h A = build_A_c2(sl);
      const int fbase = (tap * 12 + icb) * 12 + nt0;
#pragma unroll
      for (int q = 0; q < 3; ++q) {
        const v16h B = *(const v16h*)(wsBc2 + (fbase + q) * 512 + lane * 16);
        acc[q] = wmma16x16x32(A, B, acc[q]);
      }
    }
  }
#pragma unroll
  for (int q = 0; q < 3; ++q) {
    const int cn = (nt0 + q) * 16 + (lane & 15);
#pragma unroll
    for (int r = 0; r < 8; ++r)
      hout[cn * 2304 + sp0 + r + ahalf * 8] = acc[q][r];
  }
}

// ---------------------------------------------------------------------------
// Kernel 2: fused masked conv3d + MLP + posterior. Block = 4 waves = 4
// consecutive depths sharing one f16 x-slab [9][11][26]. Interior blocks pull
// the slab with ONE Tensor-Data-Mover descriptor (tensor_load_to_lds +
// s_wait_tensorcnt); boundary blocks fall back to a zero-padded manual fill.
// ---------------------------------------------------------------------------
__global__ __launch_bounds__(128) void fused_conv3d_mlp_wmma(
    const float* __restrict__ x, const _Float16* __restrict__ x16,
    const float* __restrict__ hbuf,
    const _Float16* __restrict__ wsBc3, const _Float16* __restrict__ wsBm,
    const float* __restrict__ b1, const float* __restrict__ ba,
    const float* __restrict__ bb, const float* __restrict__ bc,
    float* __restrict__ out)
{
  __shared__ __align__(16) _Float16 slab[9 * 11 * 26];   // 5.0 KB
  __shared__ __align__(16) _Float16 sbuf[4][16][96];     // per-wave activations
  __shared__ float souts[4][16][9];

  const int lane  = threadIdx.x & 31;
  const int wave  = threadIdx.x >> 5;
  const int dblk  = blockIdx.x / 144;
  const int spt   = blockIdx.x % 144;
  const int d0    = dblk * 4;
  const int d     = d0 + wave;
  const int sp0   = spt * 16;
  const int yy    = sp0 / 48;
  const int x0    = sp0 % 48;
  const int am    = lane & 15;
  const int bn    = lane & 15;
  const int ahalf = (lane >> 4) & 1;
  const int kb    = ahalf * 8;

  const bool interior = (d0 >= 5) && (yy >= 5) && (yy <= 42) && (x0 == 16);
  if (interior) {
    if (wave == 0) {
      // Tensor DMA: 3D tile 26 x 11 x 9 halfs from x16 into LDS (contiguous).
      const unsigned lds_addr =
          (unsigned)(unsigned long long)(const void*)&slab[0];
      const unsigned long long gaddr = (unsigned long long)(const void*)
          (x16 + ((d0 - 5) * 2304 + (yy - 5) * 48 + (x0 - 5)));
      v4u g0;
      g0[0] = 1u;                                   // count=1, load, user D#
      g0[1] = lds_addr;                             // lds_addr[31:0]
      g0[2] = (unsigned)(gaddr & 0xFFFFFFFFu);      // global_addr[31:0]
      g0[3] = (unsigned)((gaddr >> 32) & 0x01FFFFFFu) | (2u << 30); // type=2
      const unsigned TD = 1u << 20;                 // huge tensor dims: no clip
      v8i g1;
      g1[0] = (int)(1u << 16);                      // data_size = 2 bytes
      g1[1] = (int)((TD & 0xFFFFu) << 16);          // tensor_dim0[15:0]
      g1[2] = (int)((TD >> 16) | ((TD & 0xFFFFu) << 16)); // dim0 hi | dim1 lo
      g1[3] = (int)((TD >> 16) | (26u << 16));      // dim1 hi | tile_dim0=26
      g1[4] = (int)(11u | (9u << 16));              // tile_dim1=11 tile_dim2=9
      g1[5] = 48;                                   // tensor_dim0_stride
      g1[6] = (int)(2304u << 16);                   // dim1_stride[15:0]<<16
      g1[7] = 0;                                    // dim1_stride[47:16]=0
      v4i g2;
      g2[0] = (int)TD;                              // tensor_dim2
      g2[1] = 0; g2[2] = 0; g2[3] = 0;
      v4i g3; g3[0] = 0; g3[1] = 0; g3[2] = 0; g3[3] = 0;
#if defined(__clang_major__) && (__clang_major__ >= 23)
      v8i g4;
#pragma unroll
      for (int q = 0; q < 8; ++q) g4[q] = 0;
      __builtin_amdgcn_tensor_load_to_lds(g0, g1, g2, g3, g4, 0);
#else
      __builtin_amdgcn_tensor_load_to_lds(g0, g1, g2, g3, 0);
#endif
      __builtin_amdgcn_s_wait_tensorcnt(0);
    }
  } else {
    for (int i = threadIdx.x; i < 9 * 11 * 26; i += 128) {
      const int sd = i / 286;
      const int rr = i - sd * 286;
      const int sr = rr / 26, sc = rr - sr * 26;
      const int dz = d0 - 5 + sd, hy = yy - 5 + sr, wx = x0 - 5 + sc;
      _Float16 v = (_Float16)0.0f;
      if (dz >= 0 && hy >= 0 && hy < 48 && wx >= 0 && wx < 48)
        v = x16[dz * 2304 + hy * 48 + wx];
      slab[i] = v;
    }
  }
  __syncthreads();

  // ---- masked conv3d (mask baked into precomputed B fragments) ----
  v8f acc0, acc1;
  {
    const float bi0 = b1[bn];
    const float bi1 = (bn < 8) ? b1[bn + 16] : 0.0f;
#pragma unroll
    for (int r = 0; r < 8; ++r) { acc0[r] = bi0; acc1[r] = bi1; }
  }

#define C3STEP(KD, B, NS)                                                      \
  {                                                                            \
    const _Float16* sl = &slab[(wave + (KD)) * 286 + am];                      \
    v16h A;                                                                    \
    if (lane & 16) A = build_A_c3<(B) * 32, 8>(sl);                            \
    else           A = build_A_c3<(B) * 32, 0>(sl);                            \
    const v16h B0 = *(const v16h*)(wsBc3 + ((NS) * 2 + 0) * 512 + lane * 16);  \
    const v16h B1 = *(const v16h*)(wsBc3 + ((NS) * 2 + 1) * 512 + lane * 16);  \
    acc0 = wmma16x16x32(A, B0, acc0);                                          \
    acc1 = wmma16x16x32(A, B1, acc1);                                          \
  }

  C3STEP(0, 0, 0)  C3STEP(0, 1, 1)  C3STEP(0, 2, 2)  C3STEP(0, 3, 3)
  C3STEP(1, 0, 4)  C3STEP(1, 1, 5)  C3STEP(1, 2, 6)  C3STEP(1, 3, 7)
  C3STEP(2, 0, 8)  C3STEP(2, 1, 9)  C3STEP(2, 2, 10) C3STEP(2, 3, 11)
  C3STEP(3, 0, 12) C3STEP(3, 1, 13) C3STEP(3, 2, 14) C3STEP(3, 3, 15)
  C3STEP(4, 0, 16) C3STEP(4, 1, 17) C3STEP(4, 2, 18) C3STEP(4, 3, 19)
  C3STEP(5, 0, 20) C3STEP(5, 1, 21)
#undef C3STEP

  // ---- stage features: cols 0..23 = conv3d, col 24 = h, 25..31 = 0 ----
#pragma unroll
  for (int r = 0; r < 8; ++r) {
    const int mm = r + kb;
    sbuf[wave][mm][bn] = (_Float16)acc0[r];
    if (bn < 8) sbuf[wave][mm][bn + 16] = (_Float16)acc1[r];
  }
  if (lane < 16) {
    sbuf[wave][lane][24] = (_Float16)hbuf[d * 2304 + sp0 + lane];
#pragma unroll
    for (int cc = 25; cc < 32; ++cc) sbuf[wave][lane][cc] = (_Float16)0.0f;
  }

  const _Float16* myrow = &sbuf[wave][am][0];

  // ---- layer a: 25->48 ----
  v8f accA[3];
#pragma unroll
  for (int ntl = 0; ntl < 3; ++ntl) {
    const float bi = ba[ntl * 16 + bn];
#pragma unroll
    for (int r = 0; r < 8; ++r) accA[ntl][r] = bi;
  }
  {
    const v16h A = build_A_act<0>(myrow, kb);
#pragma unroll
    for (int ntl = 0; ntl < 3; ++ntl) {
      const v16h B = *(const v16h*)(wsBm + ntl * 512 + lane * 16);
      accA[ntl] = wmma16x16x32(A, B, accA[ntl]);
    }
  }
#pragma unroll
  for (int ntl = 0; ntl < 3; ++ntl)
#pragma unroll
    for (int r = 0; r < 8; ++r)
      sbuf[wave][r + kb][ntl * 16 + bn] = (_Float16)fmaxf(accA[ntl][r], 0.0f);
#pragma unroll
  for (int q = 0; q < 8; ++q)
    sbuf[wave][am][48 + kb + q] = (_Float16)0.0f;          // zero K-pad 48..63

  // ---- layer b: 48->96 ----
  v8f accB[6];
#pragma unroll
  for (int ntl = 0; ntl < 6; ++ntl) {
    const float bi = bb[ntl * 16 + bn];
#pragma unroll
    for (int r = 0; r < 8; ++r) accB[ntl][r] = bi;
  }
  {
    const v16h A0 = build_A_act<0>(myrow, kb);
#pragma unroll
    for (int ntl = 0; ntl < 6; ++ntl) {
      const v16h B = *(const v16h*)(wsBm + (3 + ntl) * 512 + lane * 16);
      accB[ntl] = wmma16x16x32(A0, B, accB[ntl]);
    }
    const v16h A1 = build_A_act<32>(myrow, kb);
#pragma unroll
    for (int ntl = 0; ntl < 6; ++ntl) {
      const v16h B = *(const v16h*)(wsBm + (9 + ntl) * 512 + lane * 16);
      accB[ntl] = wmma16x16x32(A1, B, accB[ntl]);
    }
  }
#pragma unroll
  for (int ntl = 0; ntl < 6; ++ntl)
#pragma unroll
    for (int r = 0; r < 8; ++r)
      sbuf[wave][r + kb][ntl * 16 + bn] = (_Float16)fmaxf(accB[ntl][r], 0.0f);

  // ---- layer c: 96->9 ----
  v8f accC;
  {
    const float bi = (bn < 9) ? bc[bn] : 0.0f;
#pragma unroll
    for (int r = 0; r < 8; ++r) accC[r] = bi;
  }
  {
    const v16h A0 = build_A_act<0>(myrow, kb);
    const v16h B0 = *(const v16h*)(wsBm + 15 * 512 + lane * 16);
    accC = wmma16x16x32(A0, B0, accC);
    const v16h A1 = build_A_act<32>(myrow, kb);
    const v16h B1 = *(const v16h*)(wsBm + 16 * 512 + lane * 16);
    accC = wmma16x16x32(A1, B1, accC);
    const v16h A2 = build_A_act<64>(myrow, kb);
    const v16h B2 = *(const v16h*)(wsBm + 17 * 512 + lane * 16);
    accC = wmma16x16x32(A2, B2, accC);
  }

  // ---- emit `out` + stash for posterior ----
#pragma unroll
  for (int r = 0; r < 8; ++r) {
    const int mm = r + kb;
    if (bn < 9) {
      const float v = accC[r];
      souts[wave][mm][bn] = v;
      out[442368 + bn * 442368 + d * 2304 + sp0 + mm] = v;
    }
  }

  // ---- posterior ----
  if (lane < 16) {
    const int sp = sp0 + lane;
    const float xv = x[d * 2304 + sp];
    float o[9];
#pragma unroll
    for (int i = 0; i < 9; ++i) o[i] = souts[wave][lane][i];
    const float mxl = fmaxf(o[6], fmaxf(o[7], o[8]));
    const float e0 = expf(o[6] - mxl), e1 = expf(o[7] - mxl),
                e2 = expf(o[8] - mxl);
    const float inv = 1.0f / (e0 + e1 + e2);
    const float ew[3] = {e0 * inv, e1 * inv, e2 * inv};
    float p = 0.0f;
#pragma unroll
    for (int i = 0; i < 3; ++i) {
      const float mu = o[i];
      float s = o[3 + i];
      s = (s == 0.0f) ? 1e-9f : s;
      s = fabsf(s);
      const float is = 1.0f / s;
      const float kk = 0.70710678118654752440f;  // 1/sqrt(2)
      const float lik = fabsf(0.5f * (erff((xv + 0.5f - mu) * is * kk) -
                                      erff((xv - 0.5f - mu) * is * kk)));
      p += ew[i] * lik;
    }
    out[d * 2304 + sp] = p;
  }
}

extern "C" void kernel_launch(void* const* d_in, const int* in_sizes, int n_in,
                              void* d_out, int out_size, void* d_ws, size_t ws_size,
                              hipStream_t stream) {
  (void)in_sizes; (void)n_in; (void)out_size; (void)ws_size;
  const float* x     = (const float*)d_in[0];
  const float* hyper = (const float*)d_in[1];
  const float* W3    = (const float*)d_in[2];
  const float* b3    = (const float*)d_in[3];
  const float* W1    = (const float*)d_in[4];
  const float* b1    = (const float*)d_in[5];
  const float* Wa    = (const float*)d_in[6];
  const float* ba    = (const float*)d_in[7];
  const float* Wb    = (const float*)d_in[8];
  const float* bb    = (const float*)d_in[9];
  const float* Wc    = (const float*)d_in[10];
  const float* bc    = (const float*)d_in[11];
  float* out = (float*)d_out;

  // workspace layout
  float*    hbuf  = (float*)d_ws;                 // 442368 f32  (1.77 MB)
  _Float16* wsBc2 = (_Float16*)(hbuf + 442368);   // 1296*512 f16 (1.33 MB)
  _Float16* wsBc3 = wsBc2 + 1296 * 512;           // 44*512 f16
  _Float16* wsBm  = wsBc3 + 44 * 512;             // 18*512 f16
  _Float16* x16   = wsBm + 18 * 512;              // 442368 f16 (0.85 MB)

  setup_Bc2 <<<(1296 * 512 + 255) / 256, 256, 0, stream>>>(W3, wsBc2);
  setup_Bc3 <<<(44 * 512 + 255) / 256, 256, 0, stream>>>(W1, wsBc3);
  setup_Bmlp<<<(18 * 512 + 255) / 256, 256, 0, stream>>>(Wa, Wb, Wc, wsBm);
  setup_x16 <<<(442368 + 255) / 256, 256, 0, stream>>>(x, x16);

  conv2d_hyper_wmma<<<144, 128, 0, stream>>>(hyper, wsBc2, b3, hbuf);
  fused_conv3d_mlp_wmma<<<6912, 128, 0, stream>>>(x, x16, hbuf, wsBc3, wsBm,
                                                  b1, ba, bb, bc, out);
}
